// GATConv_72447508349193
// MI455X (gfx1250) — matrix operand order repaired and checked by reading, not compiled, not run on hardware
//
#include <hip/hip_runtime.h>

#ifndef __has_builtin
#define __has_builtin(x) 0
#endif

#if __has_builtin(__builtin_amdgcn_tensor_load_to_lds) && \
    __has_builtin(__builtin_amdgcn_s_wait_tensorcnt)
#define USE_TDM 1
#else
#define USE_TDM 0
#endif

#if defined(__has_include)
#if __has_include(<hip/amd_detail/amd_gfx1250_TDM.h>)
#define TDM_ARGS6 1
#endif
#endif
#ifndef TDM_ARGS6
#define TDM_ARGS6 0
#endif

// Compile-time probes: these show up in hipcc stderr so the compile-only loop
// can confirm which data-movement path was selected.
#if USE_TDM
#if TDM_ARGS6
#warning "CDNA5 probe: TDM path ENABLED (tensor_load_to_lds, 6-arg form)"
#else
#warning "CDNA5 probe: TDM path ENABLED (tensor_load_to_lds, 5-arg form)"
#endif
#else
#warning "CDNA5 probe: TDM builtin NOT available - using cooperative-copy fallback"
#endif

#define N_NODES 4096
#define F_IN    512
#define F_OUT   128
#define HEADS   4
#define ALPHA   0.2f
#define NEG_INF (-9.0e15f)

typedef __bf16 bf16_t;
typedef __attribute__((ext_vector_type(16))) __bf16 v16bf;
typedef __attribute__((ext_vector_type(8)))  float  v8f;
typedef __attribute__((ext_vector_type(4)))  unsigned int u32x4;
typedef __attribute__((ext_vector_type(4)))  int  i32x4;
typedef __attribute__((ext_vector_type(8)))  int  i32x8;

// Hardware bf16 converts (v_cvt_pk_bf16_f32) instead of software bit-twiddling.
static __device__ __forceinline__ bf16_t f2bf(float f)  { return (bf16_t)f; }
static __device__ __forceinline__ float  bf2f(bf16_t b) { return (float)b; }

// ---------------------------------------------------------------------------
// Kernel 1: hT[head][o][n] = bf16( x @ W[head] + b[head] )   (WMMA bf16)
// One wave -> one 16x16 output tile. Block = 8 waves covering 16 rows x 128 cols.
// ---------------------------------------------------------------------------
__global__ __launch_bounds__(256) void gat_h_gemm(
    const float* __restrict__ x, const float* __restrict__ W,
    const float* __restrict__ b, bf16_t* __restrict__ hT) {
  const int head = blockIdx.x >> 8;          // 4 heads
  const int row0 = (blockIdx.x & 255) << 4;  // 256 row tiles
  const int col0 = (threadIdx.x >> 5) << 4;  // wave -> col tile
  const int lane = threadIdx.x & 31;
  const int half = lane >> 4;
  const int l15  = lane & 15;

  const float* xrow = x + (size_t)(row0 + l15) * F_IN;                 // A: row per lane
  const float* wcol = W + (size_t)head * F_IN * F_OUT + col0 + l15;    // B: col per lane

  v8f acc = {};
  for (int k0 = 0; k0 < F_IN; k0 += 32) {
    v16bf a, bf;
#pragma unroll
    for (int j = 0; j < 8; ++j) {                       // A 16x32 layout (ISA 7.12.2)
      a[j]     = f2bf(xrow[k0 + half * 8 + j]);
      a[8 + j] = f2bf(xrow[k0 + 16 + half * 8 + j]);
    }
#pragma unroll
    for (int e = 0; e < 16; ++e)                        // B 32x16 layout
      bf[e] = f2bf(wcol[(size_t)(k0 + half * 8 + (e & 7) + ((e >> 3) << 4)) * F_OUT]);
    acc = __builtin_amdgcn_wmma_f32_16x16x32_bf16(false, a, false, bf,
                                                  (short)0, acc, false, false);
  }
  const float bias = b[head * F_OUT + col0 + l15];
  // C layout: lane l15 = column, VGPR v = row half*8+v  -> contiguous in hT's n dim
  bf16_t* hp = hT + ((size_t)head * F_OUT + col0 + l15) * N_NODES + row0 + half * 8;
#pragma unroll
  for (int v = 0; v < 8; ++v) hp[v] = f2bf(acc[v] + bias);
}

// ---------------------------------------------------------------------------
// Kernel 2: s_src[h,n] = h[h,n,:].a_src ; s_dst[h,n] = h[h,n,:].a_dst
// ---------------------------------------------------------------------------
__global__ __launch_bounds__(256) void gat_scores(
    const bf16_t* __restrict__ hT, const float* __restrict__ a_w,
    float* __restrict__ s_src, float* __restrict__ s_dst) {
  const int t = blockIdx.x * 256 + threadIdx.x;   // t = head*4096 + n
  const int head = t >> 12;
  const bf16_t* hp = hT + (size_t)head * F_OUT * N_NODES + (t & 4095);
  const float* aw = a_w + head * 2 * F_OUT;
  float ss = 0.f, sd = 0.f;
#pragma unroll 4
  for (int o = 0; o < F_OUT; ++o) {
    float hv = bf2f(hp[(size_t)o * N_NODES]);
    ss += hv * aw[o];
    sd += hv * aw[F_OUT + o];
  }
  s_src[t] = ss;
  s_dst[t] = sd;
}

// ---------------------------------------------------------------------------
// Kernel 3: per-row softmax stats (max, 1/sum). One wave per (head,row).
// ---------------------------------------------------------------------------
__global__ __launch_bounds__(256) void gat_rowstats(
    const int* __restrict__ adj, const float* __restrict__ s_src,
    const float* __restrict__ s_dst, const float* __restrict__ a_b,
    float* __restrict__ rmax, float* __restrict__ rsinv) {
  const int gw   = blockIdx.x * 8 + (threadIdx.x >> 5);
  const int head = gw >> 12;
  const int n    = gw & 4095;
  const int lane = threadIdx.x & 31;
  const float ssrc = s_src[head * N_NODES + n];
  const float ab   = a_b[head];
  const float* sd  = s_dst + head * N_NODES;
  const int* arow  = adj + (size_t)n * N_NODES;

  float mx = -__builtin_inff();
  float sum = 0.f;
  for (int m = lane; m < N_NODES; m += 32) {
    float e = ssrc + sd[m] + ab;
    e = (e > 0.f) ? e : ALPHA * e;
    e = (arow[m] > 0) ? e : NEG_INF;
    if (e > mx) { sum = sum * __expf(mx - e) + 1.f; mx = e; }
    else        { sum += __expf(e - mx); }
  }
#pragma unroll
  for (int off = 16; off > 0; off >>= 1) {
    float mo = __shfl_xor(mx, off, 32);
    float so = __shfl_xor(sum, off, 32);
    float nm = fmaxf(mx, mo);
    sum = sum * __expf(mx - nm) + so * __expf(mo - nm);
    mx = nm;
  }
  if (lane == 0) {
    rmax[head * N_NODES + n]  = mx;
    rsinv[head * N_NODES + n] = 1.f / sum;
  }
}

// ---------------------------------------------------------------------------
// Kernel 4: out[n, head*128+o] = softmax(e) @ h   (fused, attn never stored)
// Block = 8 waves, one head; each wave owns 16 rows x 128 cols (64 acc VGPRs).
// hT slab (128x32 bf16, 8KB) staged into LDS by the Tensor Data Mover, double
// buffered so TDM(step i+1) overlaps WMMA+exp of step i. The TDM descriptor's
// pad feature (pad_interval=3 -> every 16 DWORDs, pad_amount=3 -> 4 DWORDs)
// reproduces the 80B row stride the B-fragment LDS reads expect.
// ---------------------------------------------------------------------------
__global__ __launch_bounds__(256) void gat_av(
    const int* __restrict__ adj, const bf16_t* __restrict__ hT,
    const float* __restrict__ s_src, const float* __restrict__ s_dstG,
    const float* __restrict__ a_b, const float* __restrict__ rmax,
    const float* __restrict__ rsinv, float* __restrict__ out) {
  __shared__ float  sdst_sh[N_NODES];          // 16 KB
  __shared__ bf16_t hslab[2][F_OUT * 40];      // 2 x 10 KB, stride 40 elem = 80 B

  const int head = blockIdx.x >> 5;
  const int wv   = threadIdx.x >> 5;
  const int n0   = (((blockIdx.x & 31) << 3) + wv) << 4;
  const int lane = threadIdx.x & 31;
  const int half = lane >> 4;
  const int l15  = lane & 15;
  const int n    = n0 + l15;                   // A-frag: row per lane

  for (int i = threadIdx.x; i < N_NODES; i += 256)
    sdst_sh[i] = s_dstG[head * N_NODES + i];

  const float ssrc = s_src[head * N_NODES + n];
  const float ab   = a_b[head];
  const float mrow = rmax[head * N_NODES + n];
  const float isum = rsinv[head * N_NODES + n];
  const int* arow  = adj + (size_t)n * N_NODES;

#if USE_TDM
  // D# group1 (constant): data_size=2B, pad_enable, pad_interval=3, pad_amount=3,
  // tensor_dim0=4096, tensor_dim1=128, tile_dim0=32, tile_dim1=128, dim0_stride=4096.
  const i32x8 g1 = {(int)((1u << 16) | (1u << 20) | (3u << 22) | (3u << 25)),
                    (int)0x10000000, (int)0x00800000, (int)0x00200000,
                    128, 4096, 0, 0};
#else
  const int so = threadIdx.x >> 1;             // fallback cooperative copy
  const int sm = (threadIdx.x & 1) << 4;
  const bf16_t* sgbase = hT + ((size_t)head * F_OUT + so) * N_NODES + sm;
#endif

  auto issue_slab = [&](int m0, int buf) {
#if USE_TDM
    if (threadIdx.x < 32) {                    // one wave issues the TDM op
      unsigned long long ga = (unsigned long long)(uintptr_t)hT +
          (((size_t)head * F_OUT * N_NODES) + (size_t)m0) * 2ull;
      u32x4 g0;
      g0[0] = 1u;                                        // count=1, user mode
      g0[1] = (unsigned)(uintptr_t)(&hslab[buf][0]);     // LDS byte address
      g0[2] = (unsigned)(ga & 0xffffffffull);            // global_addr[31:0]
      g0[3] = (unsigned)((ga >> 32) & 0x01fffffful) | (2u << 30);  // addr[56:32]|type=2
      i32x4 z4 = {0, 0, 0, 0};
#if TDM_ARGS6
      i32x8 z8 = {0, 0, 0, 0, 0, 0, 0, 0};
      __builtin_amdgcn_tensor_load_to_lds(g0, g1, z4, z4, z8, 0);
#else
      __builtin_amdgcn_tensor_load_to_lds(g0, g1, z4, z4, 0);
#endif
    }
#else
    const uint4* src = (const uint4*)(sgbase + m0);
    uint4* dst = (uint4*)(&hslab[buf][so * 40 + sm]);
    dst[0] = src[0];
    dst[1] = src[1];
#endif
  };

  v8f acc[8] = {};

  issue_slab(0, 0);                            // prologue: slab for step 0
  const int steps = N_NODES / 32;
  for (int i = 0; i < steps; ++i) {
    const int m0 = i << 5;
    const int cur = i & 1;
#if USE_TDM
    if (threadIdx.x < 32) __builtin_amdgcn_s_wait_tensorcnt((short)0);
#endif
    __syncthreads();                           // slab[cur] ready; slab[cur^1] free
    if (i + 1 < steps) issue_slab(m0 + 32, cur ^ 1);
    __builtin_prefetch(arow + m0 + 32, 0, 1);  // next adj chunk -> global_prefetch

    // build p as a bf16 A-fragment (lane already holds the right (n, m) pairs)
    v16bf pa;
    const int mA = m0 + half * 16;
    const int mB = m0 + half * 16 + 8;
#pragma unroll
    for (int j = 0; j < 8; ++j) {
      float e0 = ssrc + sdst_sh[mA + j] + ab;
      e0 = (e0 > 0.f) ? e0 : ALPHA * e0;
      e0 = (arow[mA + j] > 0) ? e0 : NEG_INF;
      pa[j] = f2bf(__expf(e0 - mrow) * isum);

      float e1 = ssrc + sdst_sh[mB + j] + ab;
      e1 = (e1 > 0.f) ? e1 : ALPHA * e1;
      e1 = (arow[mB + j] > 0) ? e1 : NEG_INF;
      pa[8 + j] = f2bf(__expf(e1 - mrow) * isum);
    }

#pragma unroll
    for (int ct = 0; ct < 8; ++ct) {
      union { v16bf v; uint4 q[2]; } u;
      const uint4* bp = (const uint4*)(&hslab[cur][(ct * 16 + l15) * 40 + half * 16]);
      u.q[0] = bp[0];
      u.q[1] = bp[1];
      acc[ct] = __builtin_amdgcn_wmma_f32_16x16x32_bf16(false, pa, false, u.v,
                                                        (short)0, acc[ct], false, false);
    }
  }

  // C layout: lane l15 = column, VGPR v = row half*8+v
  float* orow = out + (size_t)(n0 + half * 8) * (HEADS * F_OUT) + head * F_OUT + l15;
#pragma unroll
  for (int ct = 0; ct < 8; ++ct)
#pragma unroll
    for (int v = 0; v < 8; ++v)
      orow[(size_t)v * (HEADS * F_OUT) + ct * 16] = acc[ct][v];
}

// ---------------------------------------------------------------------------
extern "C" void kernel_launch(void* const* d_in, const int* in_sizes, int n_in,
                              void* d_out, int out_size, void* d_ws, size_t ws_size,
                              hipStream_t stream) {
  (void)in_sizes; (void)n_in; (void)out_size; (void)ws_size;
  const float* x   = (const float*)d_in[0];
  const int*   adj = (const int*)d_in[1];
  const float* W   = (const float*)d_in[2];
  const float* b   = (const float*)d_in[3];
  const float* a_w = (const float*)d_in[4];
  const float* a_b = (const float*)d_in[5];
  float* out = (float*)d_out;

  char* ws = (char*)d_ws;
  bf16_t* hT = (bf16_t*)ws;                                   // 4 MB
  size_t hT_bytes = (size_t)HEADS * F_OUT * N_NODES * sizeof(bf16_t);
  float* s_src = (float*)(ws + hT_bytes);                     // 64 KB each
  float* s_dst = s_src + HEADS * N_NODES;
  float* rmax  = s_dst + HEADS * N_NODES;
  float* rsinv = rmax + HEADS * N_NODES;

  gat_h_gemm  <<<HEADS * (N_NODES / 16),   256, 0, stream>>>(x, W, b, hT);
  gat_scores  <<<(HEADS * N_NODES) / 256,  256, 0, stream>>>(hT, a_w, s_src, s_dst);
  gat_rowstats<<<(HEADS * N_NODES) / 8,    256, 0, stream>>>(adj, s_src, s_dst, a_b, rmax, rsinv);
  gat_av      <<<HEADS * (N_NODES / 128),  256, 0, stream>>>(adj, hT, s_src, s_dst, a_b,
                                                             rmax, rsinv, out);
}
